// transformer_lm_81527069212914
// MI455X (gfx1250) — matrix operand order repaired and checked by reading, not compiled, not run on hardware
//
#include <hip/hip_runtime.h>
#include <hip/hip_bf16.h>

typedef __attribute__((ext_vector_type(16))) _Float16 v16h;
typedef __attribute__((ext_vector_type(8)))  _Float16 v8h;
typedef __attribute__((ext_vector_type(4)))  _Float16 v4h;
typedef __attribute__((ext_vector_type(8)))  float    v8f;
typedef __attribute__((ext_vector_type(4)))  unsigned int v4u;
typedef __attribute__((ext_vector_type(8)))  int      v8i;
typedef __attribute__((ext_vector_type(4)))  int      v4i;

#define WMMA_F16(a, b, c) \
  __builtin_amdgcn_wmma_f32_16x16x32_f16(false, (a), false, (b), (short)0, (c), false, false)

#define USE_TDM 1

static constexpr int Bb = 2, Nn = 2048, Cc = 1024, Hh = 16, Dd = 64;

// padded LDS column mapping: +2 DWORDs (4 halves) after every 256 DWORDs
// (512 halves).  Row stride = 1032 halves = 2064 B -> per-lane b128 reads of
// 16 different rows cover all 64 LDS banks exactly once.
__device__ __forceinline__ int padc(int c) { return c + ((c >> 9) << 2); }
static constexpr int AROW = 1032;   // halves per staged row

// ---------------------------------------------------------------------------
// Kernel 0: one-shot f32 -> f16 conversion of the two big GEMM B-matrices
// (Wqkv 3072x1024, W_out 1024x1024).  Removes all B-side cvt VALU from the
// GEMM inner loops and halves their L2 read traffic.
// ---------------------------------------------------------------------------
__global__ __launch_bounds__(512)
void cvt_weights(const float* __restrict__ Wqkv, const float* __restrict__ Wout,
                 _Float16* __restrict__ wq, _Float16* __restrict__ wo)
{
  const size_t i4 = ((size_t)blockIdx.x * 512 + threadIdx.x) * 4;
  const size_t nq = (size_t)3072 * 1024;
  const float4 v = (i4 < nq) ? *(const float4*)(Wqkv + i4)
                             : *(const float4*)(Wout + (i4 - nq));
  v4h h;
  h[0] = (_Float16)v.x; h[1] = (_Float16)v.y;
  h[2] = (_Float16)v.z; h[3] = (_Float16)v.w;
  _Float16* dst = (i4 < nq) ? (wq + i4) : (wo + (i4 - nq));
  *(v4h*)dst = h;
}

// ---------------------------------------------------------------------------
// Kernel 1: QKV projection.  Out[4096,3072] = X[4096,1024] @ Wqkv^T.
// One block per 16-row A-block: X rows converted to f16 and staged in LDS
// once (instead of 16x per-wave), each wave owns 12 N-tiles so every
// A-fragment feeds 12 WMMAs.  Results scattered to q/k [B,H,N,D] and
// v [B,H,D,N] (transposed = direct B-fragment for P@V).  1/sqrt(D) folded
// into q.
// ---------------------------------------------------------------------------
__global__ __launch_bounds__(512)
void qkv_gemm_wmma(const float* __restrict__ X, const _Float16* __restrict__ Wqf,
                   _Float16* __restrict__ qb, _Float16* __restrict__ kb,
                   _Float16* __restrict__ vb)
{
  __shared__ _Float16 Als[16 * AROW];
  const int lane = threadIdx.x & 31, wave = threadIdx.x >> 5;
  const int ln = lane & 15, hi = lane >> 4;
  const int mt = blockIdx.x;                     // 0..255

  {  // cooperative stage + convert: 512 threads x 32 contiguous floats
    const int r = threadIdx.x >> 5;
    const int c0 = (threadIdx.x & 31) * 32;
    const float* src = X + ((size_t)(mt * 16) + r) * Cc + c0;
    _Float16* dst = &Als[r * AROW + padc(c0)];
#pragma unroll
    for (int t = 0; t < 32; ++t) dst[t] = (_Float16)src[t];
  }
  __syncthreads();

  const _Float16* lrow = &Als[ln * AROW];
  v8f acc[12] = {};
  for (int kk = 0; kk < Cc; kk += 32) {
    const int ka = padc(kk);
    // A-frag per-lane K layout: {0..15} lanes 0-15, +16 for lanes 16-31
    v8h a0 = *(const v8h*)(lrow + ka + hi * 16);
    v8h a1 = *(const v8h*)(lrow + ka + 8 + hi * 16);
    v16h a;
#pragma unroll
    for (int t = 0; t < 8; ++t) { a[t] = a0[t]; a[t + 8] = a1[t]; }
#pragma unroll
    for (int t = 0; t < 12; ++t) {
      const int nt = wave * 12 + t;              // 0..191
      v16h b = *(const v16h*)(Wqf + (size_t)(nt * 16 + ln) * Cc + kk + hi * 16);
      acc[t] = WMMA_F16(a, b, acc[t]);
    }
  }

  // epilogue: C layout VGPR r -> M = r (+8 lanes 16-31), N = lane%16
#pragma unroll
  for (int t = 0; t < 12; ++t) {
    const int gc = (wave * 12 + t) * 16 + ln;    // 0..3071 = h*192 + d*3 + w
    const int h = gc / 192, rem = gc - h * 192;
    const int d = rem / 3, which = rem - d * 3;
#pragma unroll
    for (int r = 0; r < 8; ++r) {
      const int m = r + hi * 8;
      const int gr = mt * 16 + m;
      const int bi = gr >> 11, ni = gr & (Nn - 1);
      const float val = acc[t][r];
      const size_t nd = (((size_t)bi * Hh + h) * Nn + ni) * Dd + d;
      if (which == 0)      qb[nd] = (_Float16)(val * 0.125f);
      else if (which == 1) kb[nd] = (_Float16)val;
      else                 vb[(((size_t)bi * Hh + h) * Dd + d) * (size_t)Nn + ni] = (_Float16)val;
    }
  }
}

// ---------------------------------------------------------------------------
// Kernel 2: column sums of V (exact b_post handling: reference adds b_post to
// attn at ALL j incl. masked -> contributes b_post[g] * sum_j v[b,g,j,d]).
// ---------------------------------------------------------------------------
__global__ __launch_bounds__(256)
void vsum_kernel(const _Float16* __restrict__ vb, float* __restrict__ vs)
{
  __shared__ float red[256];
  const int row = blockIdx.x;                    // (b*H+h)*D+d
  const _Float16* p = vb + (size_t)row * Nn;
  float s = 0.f;
  for (int j = threadIdx.x; j < Nn; j += 256) s += (float)p[j];
  red[threadIdx.x] = s;
  __syncthreads();
  for (int st = 128; st; st >>= 1) {
    if ((int)threadIdx.x < st) red[threadIdx.x] += red[threadIdx.x + st];
    __syncthreads();
  }
  if (threadIdx.x == 0) vs[row] = red[0];
}

// ---------------------------------------------------------------------------
// Kernel 3: talking-heads causal attention, two-pass flash style.
// One workgroup = 16 waves = all 16 heads for one 16-row query block
// (the pre-softmax head mix couples heads pointwise, so all heads must be
// co-resident).  Pass 1: streaming row max/sum of mixed+biased+masked
// scores.  Pass 2: recompute, normalize, W_post-mix directly into a 16x32
// f16 A-fragment, WMMA against V.  N x N attention never touches HBM.
// ---------------------------------------------------------------------------
__global__ __launch_bounds__(512)
void attn_wmma(const _Float16* __restrict__ qb, const _Float16* __restrict__ kb,
               const _Float16* __restrict__ vb, const float* __restrict__ pos_bias,
               const float* __restrict__ W_pre, const float* __restrict__ b_pre,
               const float* __restrict__ W_post, const float* __restrict__ b_post,
               const float* __restrict__ vs, _Float16* __restrict__ ob)
{
  constexpr int LS = 34;                  // padded row stride: gcd(34,64)=2
  __shared__ float Tile[Hh * 16 * LS];    // 34 KB, reused for S then P

  const int lane = threadIdx.x & 31, g = threadIdx.x >> 5;   // g = head
  const int ln = lane & 15, hi = lane >> 4;
  const int b = blockIdx.x >> 7;
  const int ibase = (blockIdx.x & 127) << 4;

  const _Float16* qh = qb + (((size_t)b * Hh + g) * Nn) * Dd;
  const _Float16* kh = kb + (((size_t)b * Hh + g) * Nn) * Dd;
  const _Float16* vh = vb + (((size_t)b * Hh + g) * Dd) * Nn;

  // Q A-fragments for the two 32-wide d chunks (resident all kernel)
  v16h qa0, qa1;
  {
    const _Float16* q0 = qh + (size_t)(ibase + ln) * Dd;
    v8h c00 = *(const v8h*)(q0 + hi * 16);
    v8h c01 = *(const v8h*)(q0 + 8 + hi * 16);
    v8h c10 = *(const v8h*)(q0 + 32 + hi * 16);
    v8h c11 = *(const v8h*)(q0 + 40 + hi * 16);
#pragma unroll
    for (int t = 0; t < 8; ++t) {
      qa0[t] = c00[t]; qa0[t + 8] = c01[t];
      qa1[t] = c10[t]; qa1[t + 8] = c11[t];
    }
  }

  float wpre[16], wpost[16];
#pragma unroll
  for (int h = 0; h < 16; ++h) { wpre[h] = W_pre[g * 16 + h]; wpost[h] = W_post[g * 16 + h]; }
  const float bpre = b_pre[g];

  const int nsteps = (ibase + 16 + 31) >> 5;     // 32-wide causal j tiles

  float rmax[8], rsum[8];
#pragma unroll
  for (int r = 0; r < 8; ++r) { rmax[r] = -3.0e38f; rsum[r] = 0.f; }

  // raw scores for head g on [jbase,jbase+32) -> LDS -> W_pre mix for head g
  auto score_mix = [&](int jbase, float* mx0, float* mx1) {
    v8f s0 = {}, s1 = {};
    const _Float16* k0 = kh + (size_t)(jbase + ln) * Dd;
    const _Float16* k1 = kh + (size_t)(jbase + 16 + ln) * Dd;
    v16h bk;
    bk = *(const v16h*)(k0 + hi * 16);       s0 = WMMA_F16(qa0, bk, s0);
    bk = *(const v16h*)(k0 + 32 + hi * 16);  s0 = WMMA_F16(qa1, bk, s0);
    bk = *(const v16h*)(k1 + hi * 16);       s1 = WMMA_F16(qa0, bk, s1);
    bk = *(const v16h*)(k1 + 32 + hi * 16);  s1 = WMMA_F16(qa1, bk, s1);
#pragma unroll
    for (int r = 0; r < 8; ++r) {
      const int m = r + hi * 8;
      Tile[(g * 16 + m) * LS + ln]      = s0[r];
      Tile[(g * 16 + m) * LS + 16 + ln] = s1[r];
    }
    __syncthreads();
#pragma unroll
    for (int r = 0; r < 8; ++r) {
      const int m = r + hi * 8;
      float a0 = bpre, a1 = bpre;
#pragma unroll
      for (int h = 0; h < 16; ++h) {
        a0 += wpre[h] * Tile[(h * 16 + m) * LS + ln];
        a1 += wpre[h] * Tile[(h * 16 + m) * LS + 16 + ln];
      }
      const int i = ibase + m, j0 = jbase + ln, j1 = jbase + 16 + ln;
      const float* pbrow = pos_bias + ((size_t)g * Nn + i) * Nn;
      a0 += pbrow[j0];
      a1 += pbrow[j1];
      if (j0 > i) a0 = -3.0e38f;
      if (j1 > i) a1 = -3.0e38f;
      mx0[r] = a0; mx1[r] = a1;
    }
    __syncthreads();          // all S reads done before Tile is reused
  };

  // ---------------- pass 1: softmax statistics ----------------
  for (int js = 0; js < nsteps; ++js) {
    if (js + 1 < nsteps)
      __builtin_prefetch(kh + (size_t)((js + 1) * 32 + ln) * Dd, 0, 1);
    float mx0[8], mx1[8];
    score_mix(js << 5, mx0, mx1);
#pragma unroll
    for (int r = 0; r < 8; ++r) {
      float tm = fmaxf(mx0[r], mx1[r]);
#pragma unroll
      for (int sh = 8; sh >= 1; sh >>= 1) tm = fmaxf(tm, __shfl_xor(tm, sh, 32));
      const float nm = fmaxf(rmax[r], tm);
      float ts = __expf(mx0[r] - nm) + __expf(mx1[r] - nm);
#pragma unroll
      for (int sh = 8; sh >= 1; sh >>= 1) ts += __shfl_xor(ts, sh, 32);
      rsum[r] = rsum[r] * __expf(rmax[r] - nm) + ts;
      rmax[r] = nm;
    }
  }

  float rinv[8];
#pragma unroll
  for (int r = 0; r < 8; ++r) rinv[r] = (rsum[r] > 0.f) ? 1.0f / rsum[r] : 0.f;

  // ------- pass 2: recompute, normalize, post-mix, P @ V -------
  v8f acc[4] = {};
  for (int js = 0; js < nsteps; ++js) {
    const int jbase = js << 5;
    if (js + 1 < nsteps)
      __builtin_prefetch(kh + (size_t)((js + 1) * 32 + ln) * Dd, 0, 1);
    float mx0[8], mx1[8];
    score_mix(jbase, mx0, mx1);
#pragma unroll
    for (int r = 0; r < 8; ++r) {
      const int m = r + hi * 8;
      Tile[(g * 16 + m) * LS + ln]      = __expf(mx0[r] - rmax[r]) * rinv[r];
      Tile[(g * 16 + m) * LS + 16 + ln] = __expf(mx1[r] - rmax[r]) * rinv[r];
    }
    __syncthreads();
    // W_post mix built directly in 16x32 f16 A-fragment layout:
    // lanes 0-15 hold K {0..15}; lanes 16-31 hold K {16..31}
    v16h af;
#pragma unroll
    for (int t = 0; t < 16; ++t) {
      const int jl = t + hi * 16;
      float a = 0.f;
#pragma unroll
      for (int h = 0; h < 16; ++h) a += wpost[h] * Tile[(h * 16 + ln) * LS + jl];
      af[t] = (_Float16)a;
    }
    // V stored [B,H,D,N]: column d contiguous in j -> direct B-fragment
#pragma unroll
    for (int dt = 0; dt < 4; ++dt) {
      v16h bv = *(const v16h*)(vh + (size_t)(dt * 16 + ln) * Nn + jbase + hi * 16);
      acc[dt] = WMMA_F16(af, bv, acc[dt]);
    }
    __syncthreads();          // P reads done before next tile's S write
  }

  // epilogue: + b_post[g] * sum_j v ; store [B,N,H*D] f16 (A of out-proj)
  const float bpost = b_post[g];
#pragma unroll
  for (int dt = 0; dt < 4; ++dt) {
#pragma unroll
    for (int r = 0; r < 8; ++r) {
      const int m = r + hi * 8;
      const int d = dt * 16 + ln;
      const float val = acc[dt][r] + bpost * vs[((size_t)b * Hh + g) * Dd + d];
      ob[(((size_t)b * Nn + ibase + m) * Hh + g) * Dd + d] = (_Float16)val;
    }
  }
}

// ---------------------------------------------------------------------------
// Kernel 4: output projection  Y[4096,1024] = O_f16[4096,1024] @ W_out^T.
// A-block (16x1024 f16, contiguous rows) staged into LDS by the Tensor Data
// Mover: one D# per workgroup (issued by wave 0; TDM ignores EXEC), with
// pad_enable adding 2 DWORDs per 256 DWORDs so the LDS image lands in the
// same bank-conflict-free 1032-half-stride layout used everywhere else.
// ---------------------------------------------------------------------------
__global__ __launch_bounds__(512)
void out_gemm_wmma(const _Float16* __restrict__ O, const _Float16* __restrict__ Wof,
                   float* __restrict__ Y)
{
  __shared__ _Float16 Als[16 * AROW];
  const int lane = threadIdx.x & 31, wave = threadIdx.x >> 5;
  const int ln = lane & 15, hi = lane >> 4;
  const int mt = blockIdx.x;                     // 0..255

#if USE_TDM
  if (wave == 0) {
    const unsigned long long ga =
        (unsigned long long)(const void*)(O + (size_t)mt * 16 * Cc);
    const unsigned int la = (unsigned int)(size_t)(void*)&Als[0]; // LDS byte offset
    v4u g0; v8i g1;
    v4i g2 = {0, 0, 0, 0}, g3 = {0, 0, 0, 0};
    g0[0] = 1u;                                   // count=1 valid user D#
    g0[1] = la;                                   // lds_addr (bytes)
    g0[2] = (unsigned int)(ga & 0xffffffffu);     // global_addr[31:0]
    g0[3] = (unsigned int)((ga >> 32) & 0x01ffffffu) | (2u << 30); // [56:32]|type=2
    g1[0] = (1 << 16)      // data_size = 2 bytes
          | (1 << 20)      // pad_enable
          | (7 << 22)      // pad_interval: 256 DWORDs (1024 B)
          | (1 << 25);     // pad_amount: 2 DWORDs (8 B)
    g1[1] = (int)(1024u << 16);   // tensor_dim0 = 1024 elements (row length)
    g1[2] = (int)(4096u << 16);   // tensor_dim1 = 4096 rows
    g1[3] = (int)(1024u << 16);   // tile_dim0 = 1024
    g1[4] = 16;                   // tile_dim1 = 16 rows, tile_dim2 = 0
    g1[5] = 1024;                 // tensor_dim0_stride = 1024 elements
    g1[6] = 0; g1[7] = 0;
#if __clang_major__ >= 23
    v8i gx = {0, 0, 0, 0, 0, 0, 0, 0};
    __builtin_amdgcn_tensor_load_to_lds(g0, g1, g2, g3, gx, 0);
#else
    __builtin_amdgcn_tensor_load_to_lds(g0, g1, g2, g3, 0);
#endif
    __builtin_amdgcn_s_wait_tensorcnt(0);
  }
  __syncthreads();
#else
  {  // cooperative copy fallback, identical padded layout
    const int r = threadIdx.x >> 5;
    const int c0 = (threadIdx.x & 31) * 32;
    const _Float16* src = O + ((size_t)(mt * 16) + r) * Cc + c0;
    _Float16* dst = &Als[r * AROW + padc(c0)];
#pragma unroll
    for (int t = 0; t < 4; ++t)
      *(v8h*)(dst + t * 8) = *(const v8h*)(src + t * 8);
  }
  __syncthreads();
#endif

  const _Float16* lrow = &Als[ln * AROW];
  const int nt0 = wave * 4;                      // 64 N-tiles over 16 waves
  v8f acc[4] = {};
  for (int kk = 0; kk < Cc; kk += 32) {
    const int ka = padc(kk);
    v8h a0 = *(const v8h*)(lrow + ka + hi * 16);
    v8h a1 = *(const v8h*)(lrow + ka + 8 + hi * 16);
    v16h a;
#pragma unroll
    for (int t = 0; t < 8; ++t) { a[t] = a0[t]; a[t + 8] = a1[t]; }
#pragma unroll
    for (int t = 0; t < 4; ++t) {
      v16h b = *(const v16h*)(Wof + (size_t)((nt0 + t) * 16 + ln) * Cc + kk + hi * 16);
      acc[t] = WMMA_F16(a, b, acc[t]);
    }
  }
#pragma unroll
  for (int t = 0; t < 4; ++t) {
#pragma unroll
    for (int r = 0; r < 8; ++r) {
      const int m = r + hi * 8;
      Y[(size_t)(mt * 16 + m) * Cc + (nt0 + t) * 16 + ln] = acc[t][r];
    }
  }
}

// ---------------------------------------------------------------------------
extern "C" void kernel_launch(void* const* d_in, const int* in_sizes, int n_in,
                              void* d_out, int out_size, void* d_ws, size_t ws_size,
                              hipStream_t stream)
{
  const float* x     = (const float*)d_in[0];
  const float* pos   = (const float*)d_in[1];
  // d_in[2] = causal mask (bool) — causality computed analytically
  const float* Wqkv  = (const float*)d_in[3];
  const float* Wpre  = (const float*)d_in[4];
  const float* bpre  = (const float*)d_in[5];
  const float* Wpost = (const float*)d_in[6];
  const float* bpost = (const float*)d_in[7];
  const float* Wout  = (const float*)d_in[8];

  const size_t elems = (size_t)Bb * Hh * Nn * Dd;       // 4,194,304
  _Float16* qb = (_Float16*)d_ws;        // 8 MB  q [B,H,N,D] (pre-scaled)
  _Float16* kb = qb + elems;             // 8 MB  k [B,H,N,D]
  _Float16* vb = kb + elems;             // 8 MB  v [B,H,D,N] (transposed)
  _Float16* ob = vb + elems;             // 8 MB  attn out [B,N,H*D]
  float*    vsum = (float*)(ob + elems); // 8 KB  column sums of v
  _Float16* wq = (_Float16*)(vsum + Bb * Hh * Dd);      // 6 MB Wqkv f16
  _Float16* wo = wq + (size_t)3072 * 1024;              // 2 MB W_out f16

  cvt_weights<<<2048, 512, 0, stream>>>(Wqkv, Wout, wq, wo);
  qkv_gemm_wmma<<<256, 512, 0, stream>>>(x, wq, qb, kb, vb);
  vsum_kernel<<<Bb * Hh * Dd, 256, 0, stream>>>(vb, vsum);
  attn_wmma<<<Bb * (Nn / 16), 512, 0, stream>>>(qb, kb, vb, pos, Wpre, bpre,
                                                Wpost, bpost, vsum, ob);
  out_gemm_wmma<<<256, 512, 0, stream>>>(ob, wo, (float*)d_out);
}
